// Attention_11613591568639
// MI455X (gfx1250) — matrix operand order repaired and checked by reading, not compiled, not run on hardware
//
#include <hip/hip_runtime.h>
#include <hip/hip_bf16.h>

// ---------------- problem constants ----------------
#define B_   2
#define S_   2048
#define D_   1024
#define H_   16
#define DK_  64
#define BS_  (B_ * S_)      // 4096 rows for the GEMMs

// ---------------- WMMA vector types ----------------
typedef __attribute__((ext_vector_type(16))) __bf16 v16bf;
typedef __attribute__((ext_vector_type(8)))  float  v8f;

// native f32 -> bf16 conversion (hardware cvt, not manual bit-twiddle)
__device__ __forceinline__ unsigned short bf16bits(float f) {
    union { __bf16 h; unsigned short u; } v; v.h = (__bf16)f; return v.u;
}
// pack two floats as bf16 pair into one dword (v_cvt_pk_bf16_f32 class)
__device__ __forceinline__ unsigned int packbf2(float lo, float hi) {
    union { __bf16 h[2]; unsigned int u; } v;
    v.h[0] = (__bf16)lo; v.h[1] = (__bf16)hi;
    return v.u;
}
// load a v16bf fragment as two 16-byte LDS/mem chunks
__device__ __forceinline__ v16bf load_frag(const unsigned short* p0,
                                           const unsigned short* p1) {
    union { v16bf v; uint4 q[2]; } u;
    u.q[0] = *(const uint4*)p0;
    u.q[1] = *(const uint4*)p1;
    return u.v;
}
// scatter 4 scaled floats into fragment elements [base, base+4)
__device__ __forceinline__ void put4(v16bf& v, int base, float4 x, float sc) {
    v[base + 0] = (__bf16)(x.x * sc);
    v[base + 1] = (__bf16)(x.y * sc);
    v[base + 2] = (__bf16)(x.z * sc);
    v[base + 3] = (__bf16)(x.w * sc);
}

// =====================================================================
// GEMM:  C[M,N] = A[M,K] (f32) * Bw[K,N] (f32) + bias[N]
// bf16 operands, v_wmma_f32_16x16x32_bf16, f32 accumulation.
// Block tile 256x64, K-step 32, 256 threads = 8 waves.
// Wave w owns rows [w*32, w*32+32): 2 A-frags x 4 B-frags = 8 WMMA/K-step.
// LDS: A row-major, B column-major, both stride 40 halfs (80B, 16B-mult)
// so every fragment read is two aligned ds_load_b128.
// Requires M%256==0, N%64==0, K%32==0 (true for all uses here).
// =====================================================================
#define GBM 256
#define GBN 64
#define GBK 32
#define LDH 40   // LDS stride in halfs (A rows / B columns), 80B

__global__ __launch_bounds__(256) void gemm_bf16_wmma(
    const float* __restrict__ A, const float* __restrict__ Bw,
    const float* __restrict__ bias, float* __restrict__ C,
    int M, int N, int K)
{
    __shared__ __align__(16) unsigned short As[GBM * LDH]; // [row][k]
    __shared__ __align__(16) unsigned short Bs[GBN * LDH]; // [col][k]

    const int tid  = threadIdx.x;
    const int lane = tid & 31;
    const int wave = tid >> 5;          // 0..7
    const int ln   = lane & 15;
    const int lh   = lane >> 4;
    const int m0   = blockIdx.y * GBM;
    const int n0   = blockIdx.x * GBN;

    v8f acc[2][4];
    #pragma unroll
    for (int a = 0; a < 2; ++a)
        #pragma unroll
        for (int t = 0; t < 4; ++t) acc[a][t] = {};

    for (int kk = 0; kk < K; kk += GBK) {
        // ---- stage A: 256x32 f32 -> bf16, row-major ----
        #pragma unroll
        for (int c = 0; c < 8; ++c) {
            int linear = tid + c * 256;          // 0..2047 float4 chunks
            int row = linear >> 3;               // 8 float4 per row
            int col = (linear & 7) << 2;
            float4 a4 = *(const float4*)(A + (size_t)(m0 + row) * K + kk + col);
            uint2 packed;
            packed.x = packbf2(a4.x, a4.y);
            packed.y = packbf2(a4.z, a4.w);
            *(uint2*)&As[row * LDH + col] = packed;   // one ds_store_b64
        }
        // ---- stage B transposed: 32x64 f32 -> bf16, column-major ----
        {
            int rp = tid >> 4;                   // row pair 0..15 -> k = 2*rp
            int n4 = (tid & 15) << 2;            // column group
            int k  = rp << 1;
            const float* bp = Bw + (size_t)(kk + k) * N + n0 + n4;
            float4 r0 = *(const float4*)bp;
            float4 r1 = *(const float4*)(bp + N);
            unsigned int* bs32 = (unsigned int*)Bs;
            bs32[((n4 + 0) * LDH + k) >> 1] = packbf2(r0.x, r1.x);
            bs32[((n4 + 1) * LDH + k) >> 1] = packbf2(r0.y, r1.y);
            bs32[((n4 + 2) * LDH + k) >> 1] = packbf2(r0.z, r1.z);
            bs32[((n4 + 3) * LDH + k) >> 1] = packbf2(r0.w, r1.w);
        }
        __syncthreads();

        // ---- fragments: all reads are 2x ds_load_b128 ----
        v16bf af[2];
        #pragma unroll
        for (int a = 0; a < 2; ++a) {
            int row = wave * 32 + a * 16 + ln;
            af[a] = load_frag(&As[row * LDH + lh * 8],
                              &As[row * LDH + 16 + lh * 8]);
        }
        #pragma unroll
        for (int t = 0; t < 4; ++t) {
            int ncol = t * 16 + ln;
            v16bf bf = load_frag(&Bs[ncol * LDH + lh * 8],
                                 &Bs[ncol * LDH + 16 + lh * 8]);
            acc[0][t] = __builtin_amdgcn_wmma_f32_16x16x32_bf16(
                false, af[0], false, bf, (short)0, acc[0][t], false, false);
            acc[1][t] = __builtin_amdgcn_wmma_f32_16x16x32_bf16(
                false, af[1], false, bf, (short)0, acc[1][t], false, false);
        }
        __syncthreads();
    }

    // ---- epilogue: C/D layout -> global, fuse bias ----
    #pragma unroll
    for (int a = 0; a < 2; ++a)
        #pragma unroll
        for (int t = 0; t < 4; ++t)
            #pragma unroll
            for (int v = 0; v < 8; ++v) {
                int row = m0 + wave * 32 + a * 16 + v + (lh << 3);
                int col = n0 + t * 16 + ln;
                C[(size_t)row * N + col] = acc[a][t][v] + bias[col];
            }
}

// =====================================================================
// RoPE on Q and K inside the qkv buffer (rotate-half).
// =====================================================================
__global__ __launch_bounds__(256) void rope_kernel(
    float* __restrict__ qkv, const int* __restrict__ pos_ids)
{
    int idx = blockIdx.x * blockDim.x + threadIdx.x;   // < B*S*H*32 = 2^21
    int d = idx & 31;
    int h = (idx >> 5) & (H_ - 1);
    int s = (idx >> 9) & (S_ - 1);
    int b = idx >> 20;
    if (b >= B_) return;

    float pos = (float)pos_ids[b * S_ + s];
    float inv = __powf(10000.0f, -(float)d * (1.0f / 32.0f)); // BASE^(-2d/64)
    float ang = pos * inv;
    float cs = __cosf(ang), sn = __sinf(ang);

    size_t base = (size_t)(b * S_ + s) * (3u * D_) + h * DK_;
    float q1 = qkv[base + d], q2 = qkv[base + d + 32];
    qkv[base + d]      = q1 * cs - q2 * sn;
    qkv[base + d + 32] = q2 * cs + q1 * sn;

    size_t kb = base + D_;
    float k1 = qkv[kb + d], k2 = qkv[kb + d + 32];
    qkv[kb + d]      = k1 * cs - k2 * sn;
    qkv[kb + d + 32] = k2 * cs + k1 * sn;
}

// =====================================================================
// Causal flash attention, one wave per 16-row Q tile per (b,h).
// Per 32-wide K step: 4 WMMAs Q*K^T + online softmax + 4 WMMAs P*V.
// Q/K frags: direct float4 global loads (contiguous per-lane ranges).
// V: staged to LDS column-major in packed row-pairs (ds_store_b32),
//    frags via 2x ds_load_b128.
// P: C-layout -> LDS row-major -> A-layout via 2x ds_load_b128.
// =====================================================================
__global__ __launch_bounds__(32) void flash_attn_kernel(
    const float* __restrict__ qkv, float* __restrict__ out)
{
    __shared__ __align__(16) unsigned short pSh[16 * 32];   // P tile row-major
    __shared__ __align__(16) unsigned short vSh[64 * LDH];  // V tile col-major

    const int lane = threadIdx.x;
    const int ln = lane & 15, lh = lane >> 4;
    const int q0 = blockIdx.x * 16;
    const int h  = blockIdx.y;
    const int b  = blockIdx.z;

    const size_t rowStride = 3u * D_;
    const float* qBase = qkv + (size_t)b * S_ * rowStride + h * DK_;
    const float* kBase = qBase + D_;
    const float* vBase = qBase + 2 * D_;

    // ---- Q fragments (scaled by 1/sqrt(DK)) ----
    v16bf qf[2];
    {
        const float* qrow = qBase + (size_t)(q0 + ln) * rowStride;
        #pragma unroll
        for (int f = 0; f < 2; ++f) {
            const float* p = qrow + f * 32 + lh * 16;
            put4(qf[f], 0,  *(const float4*)(p + 0),  0.125f);
            put4(qf[f], 4,  *(const float4*)(p + 4),  0.125f);
            put4(qf[f], 8,  *(const float4*)(p + 8), 0.125f);
            put4(qf[f], 12, *(const float4*)(p + 12), 0.125f);
        }
    }

    v8f o[4];
    #pragma unroll
    for (int t = 0; t < 4; ++t) o[t] = {};
    float mrow[8], lrow[8];
    #pragma unroll
    for (int v = 0; v < 8; ++v) { mrow[v] = -1.0e30f; lrow[v] = 0.0f; }

    const int nsteps = (q0 + 16 + 31) >> 5;
    for (int j = 0; j < nsteps; ++j) {
        const int kc = j << 5;

        // ---- S = Q*K^T for two 16-col tiles (4 WMMAs) ----
        v8f s[2];
        #pragma unroll
        for (int t = 0; t < 2; ++t) {
            const float* krow = kBase + (size_t)(kc + t * 16 + ln) * rowStride
                              + lh * 16;
            v16bf bf0, bf1;
            put4(bf0, 0,  *(const float4*)(krow + 0),  1.0f);
            put4(bf0, 4,  *(const float4*)(krow + 4),  1.0f);
            put4(bf0, 8,  *(const float4*)(krow + 8),  1.0f);
            put4(bf0, 12, *(const float4*)(krow + 12), 1.0f);
            put4(bf1, 0,  *(const float4*)(krow + 32), 1.0f);
            put4(bf1, 4,  *(const float4*)(krow + 36), 1.0f);
            put4(bf1, 8,  *(const float4*)(krow + 40), 1.0f);
            put4(bf1, 12, *(const float4*)(krow + 44), 1.0f);
            v8f z = {};
            z = __builtin_amdgcn_wmma_f32_16x16x32_bf16(
                    false, qf[0], false, bf0, (short)0, z, false, false);
            z = __builtin_amdgcn_wmma_f32_16x16x32_bf16(
                    false, qf[1], false, bf1, (short)0, z, false, false);
            s[t] = z;
        }

        // ---- causal mask ----
        if (kc + 31 > q0) {
            #pragma unroll
            for (int t = 0; t < 2; ++t)
                #pragma unroll
                for (int v = 0; v < 8; ++v) {
                    int row = q0 + v + (lh << 3);
                    int col = kc + t * 16 + ln;
                    if (col > row) s[t][v] = -1.0e30f;
                }
        }

        // ---- online softmax (rows live in 16-lane half-groups) ----
        float rescale[8];
        #pragma unroll
        for (int v = 0; v < 8; ++v) {
            float x = fmaxf(s[0][v], s[1][v]);
            x = fmaxf(x, __shfl_xor(x, 1, 32));
            x = fmaxf(x, __shfl_xor(x, 2, 32));
            x = fmaxf(x, __shfl_xor(x, 4, 32));
            x = fmaxf(x, __shfl_xor(x, 8, 32));
            float mn = fmaxf(mrow[v], x);
            float sc = __expf(mrow[v] - mn);
            float p0 = __expf(s[0][v] - mn);
            float p1 = __expf(s[1][v] - mn);
            s[0][v] = p0; s[1][v] = p1;
            float ps = p0 + p1;
            ps += __shfl_xor(ps, 1, 32);
            ps += __shfl_xor(ps, 2, 32);
            ps += __shfl_xor(ps, 4, 32);
            ps += __shfl_xor(ps, 8, 32);
            lrow[v] = lrow[v] * sc + ps;
            mrow[v] = mn;
            rescale[v] = sc;
        }
        #pragma unroll
        for (int t = 0; t < 4; ++t)
            #pragma unroll
            for (int v = 0; v < 8; ++v) o[t][v] *= rescale[v];

        // ---- stage P (row-major) and V (column-major, packed) into LDS ----
        __syncthreads();
        #pragma unroll
        for (int v = 0; v < 8; ++v) {
            int row = v + (lh << 3);
            pSh[row * 32 + ln]      = bf16bits(s[0][v]);
            pSh[row * 32 + 16 + ln] = bf16bits(s[1][v]);
        }
        {
            unsigned int* v32 = (unsigned int*)vSh;
            #pragma unroll
            for (int r = 0; r < 32; r += 2) {
                const float* vr0 = vBase + (size_t)(kc + r) * rowStride + 2 * lane;
                float2 a  = *(const float2*)vr0;
                float2 b2 = *(const float2*)(vr0 + rowStride);
                v32[((2 * lane)     * LDH + r) >> 1] = packbf2(a.x, b2.x);
                v32[((2 * lane + 1) * LDH + r) >> 1] = packbf2(a.y, b2.y);
            }
        }
        __syncthreads();

        // ---- O += P * V (4 WMMAs) ----
        v16bf pf = load_frag(&pSh[ln * 32 + lh * 8],
                             &pSh[ln * 32 + 16 + lh * 8]);
        #pragma unroll
        for (int t = 0; t < 4; ++t) {
            int ncol = t * 16 + ln;
            v16bf vf = load_frag(&vSh[ncol * LDH + lh * 8],
                                 &vSh[ncol * LDH + 16 + lh * 8]);
            o[t] = __builtin_amdgcn_wmma_f32_16x16x32_bf16(
                       false, pf, false, vf, (short)0, o[t], false, false);
        }
    }

    // ---- normalize and write out[b, q, h*64 + dk] ----
    #pragma unroll
    for (int t = 0; t < 4; ++t)
        #pragma unroll
        for (int v = 0; v < 8; ++v) {
            int row = q0 + v + (lh << 3);
            out[((size_t)b * S_ + row) * D_ + h * DK_ + t * 16 + ln] =
                o[t][v] / lrow[v];
        }
}

// =====================================================================
// Launch: qkv GEMM -> RoPE -> flash attention -> output projection
// Workspace: qkv (BS x 3D f32 = 50.3MB) + attn (BS x D f32 = 16.8MB)
// =====================================================================
extern "C" void kernel_launch(void* const* d_in, const int* in_sizes, int n_in,
                              void* d_out, int out_size, void* d_ws, size_t ws_size,
                              hipStream_t stream) {
    const float* hs    = (const float*)d_in[0];
    const int*   pos   = (const int*)d_in[1];
    const float* wqkv  = (const float*)d_in[2];
    const float* bqkv  = (const float*)d_in[3];
    const float* wproj = (const float*)d_in[4];
    const float* bproj = (const float*)d_in[5];
    float* out = (float*)d_out;

    float* qkv  = (float*)d_ws;
    float* attn = qkv + (size_t)BS_ * 3u * D_;

    // 1) qkv = hs @ Wqkv + b      [4096 x 3072]
    gemm_bf16_wmma<<<dim3((3 * D_) / GBN, BS_ / GBM), 256, 0, stream>>>(
        hs, wqkv, bqkv, qkv, BS_, 3 * D_, D_);

    // 2) RoPE on q, k in place
    rope_kernel<<<(B_ * S_ * H_ * 32) / 256, 256, 0, stream>>>(qkv, pos);

    // 3) causal flash attention -> attn [4096 x 1024]
    flash_attn_kernel<<<dim3(S_ / 16, H_, B_), 32, 0, stream>>>(qkv, attn);

    // 4) out = attn @ Wproj + b   [4096 x 1024]
    gemm_bf16_wmma<<<dim3(D_ / GBN, BS_ / GBM), 256, 0, stream>>>(
        attn, wproj, bproj, out, BS_, D_, D_);
}
